// GraphSAGE_76613626626158
// MI455X (gfx1250) — matrix operand order repaired and checked by reading, not compiled, not run on hardware
//
#include <hip/hip_runtime.h>
#include <hip/hip_bf16.h>

#define N_NODES  100000
#define N_GRAPHS 512
#define DD       64
#define D_OUT    5
#define NTILES   (N_NODES / 16)   // 6250, exact

typedef __attribute__((ext_vector_type(16))) _Float16 v16h;
typedef __attribute__((ext_vector_type(8)))  float    v8f;

// ---------------------------------------------------------------- utilities
__global__ void zero_f32(float* __restrict__ p, int n) {
    int i = blockIdx.x * 256 + threadIdx.x;
    if (i < n) p[i] = 0.0f;
}

__global__ void degree_kernel(const int* __restrict__ dst, float* __restrict__ deg, int nE) {
    int e = blockIdx.x * 256 + threadIdx.x;
    if (e < nE) atomicAdd(&deg[dst[e]], 1.0f);
}

__global__ void graph_count_kernel(const int* __restrict__ batch, float* __restrict__ cnt) {
    int n = blockIdx.x * 256 + threadIdx.x;
    if (n < N_NODES) atomicAdd(&cnt[batch[n]], 1.0f);
}

// agg[dst] += feat[src]; one thread handles one float4 chunk of one edge
__global__ void scatter_kernel(const float* __restrict__ feat,
                               const int* __restrict__ src,
                               const int* __restrict__ dst,
                               float* __restrict__ agg, int nE) {
    int tid = blockIdx.x * 256 + threadIdx.x;
    if (tid >= nE * 16) return;
    int e = tid >> 4;
    int c = tid & 15;
    int s = src[e];
    int d = dst[e];
    float4 v = ((const float4*)(feat + (long long)s * DD))[c];
    float* o = agg + (long long)d * DD + c * 4;
    atomicAdd(o + 0, v.x);
    atomicAdd(o + 1, v.y);
    atomicAdd(o + 2, v.z);
    atomicAdd(o + 3, v.w);
}

// ---------------------------------------------------------------- WMMA SAGE layer
// out = relu( (agg/max(deg,1)) @ Wl^T + bl + xin @ Wr^T ), all [N_NODES, 64]
// One wave per 16-row tile; 8 waves per 256-thread block.
// Weights are pulled into LDS with GLOBAL_LOAD_ASYNC_TO_LDS_B128 (ASYNCcnt),
// then converted once to f16 for the WMMA B fragments.
__global__ __launch_bounds__(256)
void sage_layer_wmma(const float* __restrict__ xin,
                     const float*              agg,     // may alias out
                     const float* __restrict__ deg,
                     const float* __restrict__ Wl,
                     const float* __restrict__ bl,
                     const float* __restrict__ Wr,
                     float*                    out) {
    __shared__ float    stg[2 * DD * DD];   // fp32 staging: Wl then Wr (32 KB)
    __shared__ _Float16 sWl[DD * DD];       // f16 tiles for B fragments (8 KB each)
    __shared__ _Float16 sWr[DD * DD];

    // --- async copy of both weight matrices into LDS (16B chunks per lane) ---
    {
        const unsigned ldsBase = (unsigned)(size_t)(&stg[0]);   // low 32b = LDS offset
        for (int i = threadIdx.x; i < (DD * DD * 4) / 16; i += 256) {  // 1024 chunks
            unsigned lofsL = ldsBase + (unsigned)i * 16u;
            unsigned lofsR = ldsBase + (unsigned)(DD * DD * 4) + (unsigned)i * 16u;
            unsigned gofs  = (unsigned)i * 16u;
            asm volatile("global_load_async_to_lds_b128 %0, %1, %2"
                         :: "v"(lofsL), "v"(gofs), "s"(Wl) : "memory");
            asm volatile("global_load_async_to_lds_b128 %0, %1, %2"
                         :: "v"(lofsR), "v"(gofs), "s"(Wr) : "memory");
        }
        asm volatile("s_wait_asynccnt 0" ::: "memory");
        __syncthreads();
        for (int i = threadIdx.x; i < DD * DD; i += 256) {
            sWl[i] = (_Float16)stg[i];
            sWr[i] = (_Float16)stg[DD * DD + i];
        }
        __syncthreads();
    }

    const int wave = threadIdx.x >> 5;
    const int lane = threadIdx.x & 31;
    const int tile = blockIdx.x * 8 + wave;
    if (tile >= NTILES) return;           // wave-uniform: EXEC stays all-1s

    const int lrow = lane & 15;           // row within tile (A,M) / column (B,C,D, N)
    const int kgrp = lane >> 4;           // K half selector
    const int row  = tile * 16 + lrow;

    const float rnorm = 1.0f / fmaxf(deg[row], 1.0f);

    // A fragments (16x32 f16 layout): lanes 0-15 -> K = kb..kb+7, kb+16..kb+23 (kb = s*32)
    //                                 lanes 16-31 -> +8
    const float* aggRow = agg + (long long)row * DD;
    const float* xRow   = xin + (long long)row * DD;
    v16h aA[2], aX[2];
    #pragma unroll
    for (int s = 0; s < 2; ++s) {
        const int kb = s * 32 + kgrp * 8;
        #pragma unroll
        for (int e = 0; e < 8; ++e) {
            aA[s][e]     = (_Float16)(aggRow[kb + e]      * rnorm);
            aA[s][e + 8] = (_Float16)(aggRow[kb + 16 + e] * rnorm);
            aX[s][e]     = (_Float16)(xRow[kb + e]);
            aX[s][e + 8] = (_Float16)(xRow[kb + 16 + e]);
        }
    }

    // 4 output N-tiles of 16 columns each
    #pragma unroll
    for (int nt = 0; nt < 4; ++nt) {
        const int n = nt * 16 + lrow;
        const float bias = bl[n];
        v8f c;
        #pragma unroll
        for (int i = 0; i < 8; ++i) c[i] = bias;   // bias depends only on column n

        #pragma unroll
        for (int s = 0; s < 2; ++s) {
            // B fragment (32x16): lane holds column n, K = s*32 + kgrp*8 + (e&7) + 16*(e>>3)
            const int kb = s * 32 + kgrp * 8;
            v16h bL, bR;
            #pragma unroll
            for (int e = 0; e < 16; ++e) {
                bL[e] = sWl[n * DD + kb + (e & 7) + ((e >> 3) << 4)];
                bR[e] = sWr[n * DD + kb + (e & 7) + ((e >> 3) << 4)];
            }
            c = __builtin_amdgcn_wmma_f32_16x16x32_f16(false, aA[s], false, bL,
                                                       (short)0, c, false, false);
            c = __builtin_amdgcn_wmma_f32_16x16x32_f16(false, aX[s], false, bR,
                                                       (short)0, c, false, false);
        }

        // D layout: VGPR i, lanes 0-15 -> M=i ; lanes 16-31 -> M=i+8 ; N = lane&15
        #pragma unroll
        for (int i = 0; i < 8; ++i) {
            const int m = tile * 16 + i + 8 * kgrp;
            out[(long long)m * DD + n] = fmaxf(c[i], 0.0f);
        }
    }
}

// ---------------------------------------------------------------- pooling + head
__global__ void pool_kernel(const float* __restrict__ h,
                            const int* __restrict__ batch,
                            float* __restrict__ gsum) {
    int tid = blockIdx.x * 256 + threadIdx.x;
    if (tid >= N_NODES * 16) return;
    int n = tid >> 4;
    int c = tid & 15;
    int g = batch[n];
    float4 v = ((const float4*)(h + (long long)n * DD))[c];
    float* o = gsum + (long long)g * DD + c * 4;
    atomicAdd(o + 0, v.x);
    atomicAdd(o + 1, v.y);
    atomicAdd(o + 2, v.z);
    atomicAdd(o + 3, v.w);
}

// one block (64 threads) per graph: out = Wo @ relu(Wg @ (gsum/cnt) + bg) + bo
__global__ __launch_bounds__(64)
void head_kernel(const float* __restrict__ gsum,
                 const float* __restrict__ cnt,
                 const float* __restrict__ Wg, const float* __restrict__ bg,
                 const float* __restrict__ Wo, const float* __restrict__ bo,
                 float* __restrict__ outp) {
    __shared__ float gmean[DD];
    __shared__ float hid[DD];
    const int g = blockIdx.x;
    const int t = threadIdx.x;
    const float c = fmaxf(cnt[g], 1.0f);
    gmean[t] = gsum[g * DD + t] / c;
    __syncthreads();
    float acc = bg[t];
    #pragma unroll 8
    for (int k = 0; k < DD; ++k) acc += Wg[t * DD + k] * gmean[k];
    hid[t] = fmaxf(acc, 0.0f);
    __syncthreads();
    if (t < D_OUT) {
        float o = bo[t];
        #pragma unroll 8
        for (int k = 0; k < DD; ++k) o += Wo[t * DD + k] * hid[k];
        outp[g * D_OUT + t] = o;
    }
}

// ---------------------------------------------------------------- launcher
extern "C" void kernel_launch(void* const* d_in, const int* in_sizes, int n_in,
                              void* d_out, int out_size, void* d_ws, size_t ws_size,
                              hipStream_t stream) {
    const float* x   = (const float*)d_in[0];
    const float* W1l = (const float*)d_in[1];
    const float* b1l = (const float*)d_in[2];
    const float* W1r = (const float*)d_in[3];
    const float* W2l = (const float*)d_in[4];
    const float* b2l = (const float*)d_in[5];
    const float* W2r = (const float*)d_in[6];
    const float* Wg  = (const float*)d_in[7];
    const float* bg  = (const float*)d_in[8];
    const float* Wo  = (const float*)d_in[9];
    const float* bo  = (const float*)d_in[10];
    const int*   ei  = (const int*)d_in[11];
    const int*   batch = (const int*)d_in[12];
    float*       outp  = (float*)d_out;

    const int nE  = in_sizes[11] / 2;      // 1,600,000
    const int* src = ei;
    const int* dst = ei + nE;

    // workspace layout (floats), all offsets 64B-aligned
    float* ws    = (float*)d_ws;
    float* deg   = ws;                                  // N_NODES
    float* cnt   = deg  + N_NODES;                      // N_GRAPHS
    float* gsum  = cnt  + N_GRAPHS;                     // N_GRAPHS*64
    float* agg   = gsum + (size_t)N_GRAPHS * DD;        // N_NODES*64 (also h2)
    float* h1    = agg  + (size_t)N_NODES * DD;         // N_NODES*64

    const int zero1 = N_NODES + N_GRAPHS + N_GRAPHS * DD + N_NODES * DD;

    // 1) zero deg/cnt/gsum/agg
    zero_f32<<<(zero1 + 255) / 256, 256, 0, stream>>>(deg, zero1);

    // 2) degrees and per-graph counts
    degree_kernel<<<(nE + 255) / 256, 256, 0, stream>>>(dst, deg, nE);
    graph_count_kernel<<<(N_NODES + 255) / 256, 256, 0, stream>>>(batch, cnt);

    // 3) layer-1 scatter + fused SAGE GEMM (WMMA)
    scatter_kernel<<<(nE * 16 + 255) / 256, 256, 0, stream>>>(x, src, dst, agg, nE);
    sage_layer_wmma<<<(NTILES + 7) / 8, 256, 0, stream>>>(x, agg, deg, W1l, b1l, W1r, h1);

    // 4) layer-2 scatter + fused SAGE GEMM, output in-place over agg
    zero_f32<<<((N_NODES * DD) + 255) / 256, 256, 0, stream>>>(agg, N_NODES * DD);
    scatter_kernel<<<(nE * 16 + 255) / 256, 256, 0, stream>>>(h1, src, dst, agg, nE);
    sage_layer_wmma<<<(NTILES + 7) / 8, 256, 0, stream>>>(h1, agg, deg, W2l, b2l, W2r, agg);

    // 5) global mean pool + MLP head
    pool_kernel<<<(N_NODES * 16 + 255) / 256, 256, 0, stream>>>(agg, batch, gsum);
    head_kernel<<<N_GRAPHS, 64, 0, stream>>>(gsum, cnt, Wg, bg, Wo, bo, outp);
}
